// TransformerTransducer_73280732004784
// MI455X (gfx1250) — matrix operand order repaired and checked by reading, not compiled, not run on hardware
//
#include <hip/hip_runtime.h>
#include <hip/hip_bf16.h>
#include <math.h>

// Problem constants (from reference)
#define BB 4
#define MM 256
#define NN 64
#define DD 512
#define CC 1024

typedef _Float16 v4h  __attribute__((ext_vector_type(4)));
typedef _Float16 v8h  __attribute__((ext_vector_type(8)));
typedef _Float16 v16h __attribute__((ext_vector_type(16)));
typedef float    v8f  __attribute__((ext_vector_type(8)));

union Frag16 { v16h v; v8h h[2]; };

// ---------------------------------------------------------------------------
// Kernel A: fp32 -> f16 conversion (elementwise)
// ---------------------------------------------------------------------------
__global__ void cvt_f32_to_f16(const float* __restrict__ in,
                               _Float16* __restrict__ out, int n) {
    int i = blockIdx.x * blockDim.x + threadIdx.x;
    if (i < n) out[i] = (_Float16)in[i];
}

// ---------------------------------------------------------------------------
// Kernel B: projections Penc = enc @ W1[:, :D]^T ; Pdec = dec @ W1[:, D:]^T
// Rows 0..1023 -> enc side, rows 1024..1279 -> dec side. 16 rows/block,
// 8 waves x 4 c-tiles = 512 output columns. WMMA f32_16x16x32_f16, K=512.
// ---------------------------------------------------------------------------
__global__ __launch_bounds__(256)
void proj_kernel(const _Float16* __restrict__ ench,
                 const _Float16* __restrict__ dech,
                 const _Float16* __restrict__ W1h,   // [512][1024] row-major
                 float* __restrict__ Penc,           // [1024][512]
                 float* __restrict__ Pdec)           // [256][512]
{
    const int row0  = blockIdx.x * 16;          // 0..1279
    const bool isDec = (row0 >= BB * MM);
    const _Float16* src = isDec ? dech + (size_t)(row0 - BB * MM) * DD
                                : ench + (size_t)row0 * DD;
    const int dofs = isDec ? DD : 0;

    const int tid  = threadIdx.x;
    const int wave = tid >> 5;
    const int lane = tid & 31;
    const int l15  = lane & 15;
    const int hi   = lane >> 4;                 // 0: lanes 0-15, 1: lanes 16-31
    const int cbase = wave * 64;

    v8f acc[4] = {};

    for (int k0 = 0; k0 < DD; k0 += 32) {
        // A fragment: 16 rows x 32 K (f16). lane<16: K {0..7,16..23}; lane>=16: +8
        Frag16 a;
        const _Float16* ap = src + (size_t)l15 * DD + k0 + hi * 8;
        a.h[0] = *(const v8h*)ap;
        a.h[1] = *(const v8h*)(ap + 16);
#pragma unroll
        for (int j = 0; j < 4; ++j) {
            const int c = cbase + j * 16 + l15;     // output channel (N dim)
            const _Float16* bp = W1h + (size_t)c * (2 * DD) + dofs + k0 + hi * 8;
            Frag16 bf;
            bf.h[0] = *(const v8h*)bp;
            bf.h[1] = *(const v8h*)(bp + 16);
            acc[j] = __builtin_amdgcn_wmma_f32_16x16x32_f16(
                false, a.v, false, bf.v, (short)0, acc[j], false, false);
        }
    }

    float* P = isDec ? Pdec : Penc;
    const int prow0 = isDec ? row0 - BB * MM : row0;
#pragma unroll
    for (int j = 0; j < 4; ++j) {
        const int c = cbase + j * 16 + l15;
#pragma unroll
        for (int g = 0; g < 8; ++g) {
            const int pr = prow0 + g + hi * 8;      // C/D layout: M = g (+8 hi half)
            P[(size_t)pr * DD + c] = acc[j][g];
        }
    }
}

// ---------------------------------------------------------------------------
// Kernel C: fused joint network.
// Block = one (b,m) x ALL 64 n rows (maximizes W2 reuse: each B fragment
// feeds 4 WMMAs). 512 threads = 16 waves; wave w owns cols [64w, 64w+64)
// for all 4 row-tiles.
//   Phase 1: hidden = tanh(Penc[m] + Pdec[n] + b1) -> f16 in LDS (stride 520h)
//   Phase 2: logits = hidden @ W2h^T + b2 via WMMA f32_16x16x32_f16
//   Phase 3: row-wise log_softmax fused in registers (shfl_xor + LDS reduce).
// ---------------------------------------------------------------------------
#define HSTR 520  // LDS hidden stride in halfs: 1040B = 65*16B -> aligned, no bank conflicts

__global__ __launch_bounds__(512)
void joint_kernel(const float* __restrict__ Penc,   // [B*M][D]
                  const float* __restrict__ Pdec,   // [B*N][D]
                  const float* __restrict__ b1,     // [D]
                  const _Float16* __restrict__ W2h, // [C][D] row-major f16
                  const float* __restrict__ b2,     // [C]
                  float* __restrict__ out)          // [B,M,N,C]
{
    __shared__ _Float16 sh_h[64 * HSTR];   // 66,560 B
    __shared__ float sh_pmax[16][64];      // 4 KB
    __shared__ float sh_psum[16][64];      // 4 KB
    __shared__ float sh_max[64];
    __shared__ float sh_ls[64];

    const int bm  = blockIdx.x;          // b*M + m, 0..1023
    const int b   = bm >> 8;             // / M
    const int tid = threadIdx.x;

    // ---- Phase 1: hidden tile (64 rows x 512) into LDS as f16, float4-vectorized ----
    const float* pe  = Penc + (size_t)bm * DD;
    const float* pd0 = Pdec + (size_t)(b * NN) * DD;
    for (int i = tid; i < NN * (DD / 4); i += 512) {
        const int r = i >> 7;                 // / (DD/4)
        const int d = (i & 127) << 2;
        const float4 pv = *(const float4*)(pe + d);
        const float4 dv = *(const float4*)(pd0 + (size_t)r * DD + d);
        const float4 bv = *(const float4*)(b1 + d);
        v4h h;
        h.x = (_Float16)tanhf(pv.x + dv.x + bv.x);
        h.y = (_Float16)tanhf(pv.y + dv.y + bv.y);
        h.z = (_Float16)tanhf(pv.z + dv.z + bv.z);
        h.w = (_Float16)tanhf(pv.w + dv.w + bv.w);
        *(v4h*)&sh_h[r * HSTR + d] = h;
    }
    __syncthreads();

    // ---- Phase 2: WMMA GEMM ----
    const int wave = tid >> 5;           // 0..15
    const int lane = tid & 31;
    const int l15  = lane & 15;
    const int hi   = lane >> 4;
    const int cbase = wave * 64;

    v8f acc[4][4] = {};   // [row-tile][c-tile]

    for (int k0 = 0; k0 < DD; k0 += 32) {
        Frag16 a[4];
#pragma unroll
        for (int rt = 0; rt < 4; ++rt) {
            const _Float16* ap = &sh_h[(rt * 16 + l15) * HSTR + k0 + hi * 8];
            a[rt].h[0] = *(const v8h*)ap;         // ds_load_b128
            a[rt].h[1] = *(const v8h*)(ap + 16);  // ds_load_b128
        }
#pragma unroll
        for (int j = 0; j < 4; ++j) {
            const int c = cbase + j * 16 + l15;
            const _Float16* bp = W2h + (size_t)c * DD + k0 + hi * 8;
            Frag16 bf;
            bf.h[0] = *(const v8h*)bp;            // global_load_b128 (L2-resident)
            bf.h[1] = *(const v8h*)(bp + 16);
#pragma unroll
            for (int rt = 0; rt < 4; ++rt)
                acc[rt][j] = __builtin_amdgcn_wmma_f32_16x16x32_f16(
                    false, a[rt].v, false, bf.v, (short)0, acc[rt][j], false, false);
        }
    }

    // ---- add b2 (per-column bias; column fixed per lane across all g) ----
    float bb[4];
#pragma unroll
    for (int j = 0; j < 4; ++j) bb[j] = b2[cbase + j * 16 + l15];
#pragma unroll
    for (int rt = 0; rt < 4; ++rt)
#pragma unroll
        for (int j = 0; j < 4; ++j)
#pragma unroll
            for (int g = 0; g < 8; ++g) acc[rt][j][g] += bb[j];

    // ---- Phase 3a: per-row max (wave-local 64 cols -> cross-wave via LDS) ----
#pragma unroll
    for (int rt = 0; rt < 4; ++rt)
#pragma unroll
        for (int g = 0; g < 8; ++g) {
            float pm = acc[rt][0][g];
            pm = fmaxf(pm, acc[rt][1][g]);
            pm = fmaxf(pm, acc[rt][2][g]);
            pm = fmaxf(pm, acc[rt][3][g]);
            pm = fmaxf(pm, __shfl_xor(pm, 1, 32));
            pm = fmaxf(pm, __shfl_xor(pm, 2, 32));
            pm = fmaxf(pm, __shfl_xor(pm, 4, 32));
            pm = fmaxf(pm, __shfl_xor(pm, 8, 32));
            if (l15 == 0) sh_pmax[wave][rt * 16 + g + hi * 8] = pm;
        }
    __syncthreads();
    if (tid < 64) {
        float m = sh_pmax[0][tid];
#pragma unroll
        for (int w = 1; w < 16; ++w) m = fmaxf(m, sh_pmax[w][tid]);
        sh_max[tid] = m;
    }
    __syncthreads();

    // ---- Phase 3b: per-row sum of exp ----
#pragma unroll
    for (int rt = 0; rt < 4; ++rt)
#pragma unroll
        for (int g = 0; g < 8; ++g) {
            const float mx = sh_max[rt * 16 + g + hi * 8];
            float s = __expf(acc[rt][0][g] - mx) + __expf(acc[rt][1][g] - mx)
                    + __expf(acc[rt][2][g] - mx) + __expf(acc[rt][3][g] - mx);
            s += __shfl_xor(s, 1, 32);
            s += __shfl_xor(s, 2, 32);
            s += __shfl_xor(s, 4, 32);
            s += __shfl_xor(s, 8, 32);
            if (l15 == 0) sh_psum[wave][rt * 16 + g + hi * 8] = s;
        }
    __syncthreads();
    if (tid < 64) {
        float s = sh_psum[0][tid];
#pragma unroll
        for (int w = 1; w < 16; ++w) s += sh_psum[w][tid];
        sh_ls[tid] = __logf(s);
    }
    __syncthreads();

    // ---- Phase 3c: write out = x - max - log(sum) ----
    const size_t outbase = (size_t)bm * NN * CC;
#pragma unroll
    for (int rt = 0; rt < 4; ++rt)
#pragma unroll
        for (int g = 0; g < 8; ++g) {
            const int row = rt * 16 + g + hi * 8;
            const float sub = sh_max[row] + sh_ls[row];
            float* op = out + outbase + (size_t)row * CC;
#pragma unroll
            for (int j = 0; j < 4; ++j)
                op[cbase + j * 16 + l15] = acc[rt][j][g] - sub;
        }
}

// ---------------------------------------------------------------------------
// Launch
// ---------------------------------------------------------------------------
extern "C" void kernel_launch(void* const* d_in, const int* in_sizes, int n_in,
                              void* d_out, int out_size, void* d_ws, size_t ws_size,
                              hipStream_t stream) {
    const float* enc = (const float*)d_in[0];  // [4,256,512]
    const float* dec = (const float*)d_in[1];  // [4,64,512]
    const float* W1  = (const float*)d_in[2];  // [512,1024]
    const float* b1  = (const float*)d_in[3];  // [512]
    const float* W2  = (const float*)d_in[4];  // [1024,512]
    const float* b2  = (const float*)d_in[5];  // [1024]
    float* out = (float*)d_out;

    char* ws = (char*)d_ws;
    _Float16* ench = (_Float16*)(ws + 0);            // 1,048,576 B
    _Float16* dech = (_Float16*)(ws + 1048576);      //   262,144 B
    _Float16* W1h  = (_Float16*)(ws + 1310720);      // 1,048,576 B
    _Float16* W2h  = (_Float16*)(ws + 2359296);      // 1,048,576 B
    float*    Penc = (float*)(ws + 3407872);         // 2,097,152 B
    float*    Pdec = (float*)(ws + 5505024);         //   524,288 B  (total ~5.75 MB)

    cvt_f32_to_f16<<<(BB*MM*DD)/256, 256, 0, stream>>>(enc, ench, BB*MM*DD);
    cvt_f32_to_f16<<<(BB*NN*DD)/256, 256, 0, stream>>>(dec, dech, BB*NN*DD);
    cvt_f32_to_f16<<<(DD*2*DD)/256, 256, 0, stream>>>(W1, W1h, DD*2*DD);
    cvt_f32_to_f16<<<(CC*DD)/256, 256, 0, stream>>>(W2, W2h, CC*DD);

    proj_kernel<<<(BB*MM + BB*NN)/16, 256, 0, stream>>>(ench, dech, W1h, Penc, Pdec);

    joint_kernel<<<BB*MM, 512, 0, stream>>>(Penc, Pdec, b1, W2h, b2, out);
}
